// LinearAttention3D_66125316489928
// MI455X (gfx1250) — matrix operand order repaired and checked by reading, not compiled, not run on hardware
//
#include <hip/hip_runtime.h>
#include <hip/hip_bf16.h>

#define BATCH 8
#define CH    256
#define SP    16384   // H*W
#define NHD   8
#define HD    32

typedef __attribute__((ext_vector_type(16))) __bf16 v16bf;
typedef __attribute__((ext_vector_type(8)))  float  v8f;
typedef unsigned int u32x4 __attribute__((ext_vector_type(4)));
typedef int          i32x4 __attribute__((ext_vector_type(4)));
typedef int          i32x8 __attribute__((ext_vector_type(8)));

#if defined(__has_builtin)
#if __has_builtin(__builtin_amdgcn_tensor_load_to_lds)
#define HAVE_TDM 1
#endif
#if __has_builtin(__builtin_amdgcn_global_load_async_to_lds_b32)
#define HAVE_ASYNC 1
#endif
#endif
#ifndef HAVE_TDM
#define HAVE_TDM 0
#endif
#ifndef HAVE_ASYNC
#define HAVE_ASYNC 0
#endif

// Builtin wants a (non-const) AS1 int* source and an AS3 int* LDS destination.
// Go through integers so const/type qualifiers never matter.
typedef __attribute__((address_space(1))) int* as1i;
typedef __attribute__((address_space(3))) int* as3i;
#define TO_AS1(p) ((as1i)(unsigned long long)(uintptr_t)(p))
#define TO_AS3(p) ((as3i)(uintptr_t)(unsigned)(uintptr_t)(p))

union FragB { v16bf v; unsigned int u[8]; };
union AccF  { v8f v; float f[8]; };

__device__ __forceinline__ unsigned short f2bf(float f) {
    union { float f; unsigned int u; } x; x.f = f;
    unsigned int r = x.u + 0x7FFFu + ((x.u >> 16) & 1u);
    return (unsigned short)(r >> 16);
}
__device__ __forceinline__ float bf2f(unsigned short h) {
    union { unsigned int u; float f; } x; x.u = ((unsigned int)h) << 16;
    return x.f;
}
__device__ __forceinline__ v8f zero_v8f() {
    v8f z;
#pragma unroll
    for (int r = 0; r < 8; ++r) z[r] = 0.0f;
    return z;
}
__device__ __forceinline__ void wait_tensorcnt0() {
#if defined(__has_builtin) && __has_builtin(__builtin_amdgcn_s_wait_tensorcnt)
    __builtin_amdgcn_s_wait_tensorcnt(0);
#else
    asm volatile("s_wait_tensorcnt 0x0" ::: "memory");
#endif
}
__device__ __forceinline__ void wait_asynccnt0() {
#if defined(__has_builtin) && __has_builtin(__builtin_amdgcn_s_wait_asynccnt)
    __builtin_amdgcn_s_wait_asynccnt(0);
#else
    asm volatile("s_wait_asynccnt 0x0" ::: "memory");
#endif
}

// A fragment: 16x32 bf16, M rows x K cols, from LDS row-major [M][ld]
// lanes 0-15 row M=lane, K=0..15; lanes 16-31 K=16..31 (two K per VGPR)
__device__ __forceinline__ v16bf load_frag_a(const unsigned short* lds, int ld, int mbase) {
    const int lane = threadIdx.x & 31;
    const int row  = mbase + (lane & 15);
    const int klo  = (lane < 16) ? 0 : 16;
    FragB f;
#pragma unroll
    for (int v = 0; v < 8; ++v) {
        int k = klo + v * 2;
        f.u[v] = *(const unsigned int*)(lds + row * ld + k);
    }
    return f.v;
}

// B fragment: 32x16 bf16, K rows x N cols, from LDS stored transposed [N][ld]
// lanes 0-15: col N=lane, K=0..15; lanes 16-31: K=16..31 (two K per VGPR)
__device__ __forceinline__ v16bf load_frag_b(const unsigned short* lds, int ld, int nbase) {
    const int lane = threadIdx.x & 31;
    const int col  = nbase + (lane & 15);
    const int klo  = (lane < 16) ? 0 : 16;
    FragB f;
#pragma unroll
    for (int v = 0; v < 8; ++v) {
        int k = klo + v * 2;
        f.u[v] = *(const unsigned int*)(lds + col * ld + k);
    }
    return f.v;
}

// ---------------- convert / init ----------------
__global__ void f32_to_bf16_kernel(const float* __restrict__ in,
                                   unsigned short* __restrict__ out, size_t n) {
    size_t i = ((size_t)blockIdx.x * blockDim.x + threadIdx.x) * 4;
    if (i + 3 < n) {
        float4 v = *(const float4*)(in + i);
        unsigned int p0 = (unsigned int)f2bf(v.x) | ((unsigned int)f2bf(v.y) << 16);
        unsigned int p1 = (unsigned int)f2bf(v.z) | ((unsigned int)f2bf(v.w) << 16);
        *(unsigned int*)(out + i)     = p0;
        *(unsigned int*)(out + i + 2) = p1;
    }
}

__global__ void zero_f32_kernel(float* __restrict__ p, size_t n) {
    size_t i = (size_t)blockIdx.x * blockDim.x + threadIdx.x;
    if (i < n) p[i] = 0.0f;
}

// ---------------- projection GEMM: out[o,n] = sum_c W[o,c]*X[c,n] + bias[o] ----------------
// Block tile 128M x 128N, K-step 32. A tile DMA'd by the Tensor Data Mover with
// hardware LDS padding -> [128][36] u16 layout; B tile transposed by threads.
// MODE 0: store bf16 (qp). MODE 1: exp + store bf16 + row-sum atomics (ekp).
// MODE 2: store bf16 (vp). MODE 3: store f32 to d_out (final proj).
template <int MODE>
__global__ void proj_gemm_kernel(const unsigned short* __restrict__ X,   // [B][CH][SP] bf16
                                 const unsigned short* __restrict__ Wm,  // [CH][CH] bf16
                                 const float* __restrict__ bias,         // [CH]
                                 unsigned short* __restrict__ outb,
                                 float* __restrict__ outf,
                                 float* __restrict__ ksum) {
    __shared__ __align__(16) unsigned short ldsA[128 * 36];  // 9216 B (TDM-padded rows)
    __shared__ __align__(16) unsigned short ldsB[128 * 34];  // 8704 B, [N][K] transposed
    const int t    = threadIdx.x;
    const int lane = t & 31;
    const int wave = t >> 5;
    const int nblk = blockIdx.x;   // SP/128 tiles
    const int mblk = blockIdx.y;   // CH/128 tiles
    const int b    = blockIdx.z;
    const unsigned short* Xb = X + (size_t)b * CH * SP;
    const unsigned short* Wt = Wm + (size_t)(mblk * 128) * CH;  // A panel origin

    AccF acc[8];
#pragma unroll
    for (int i = 0; i < 8; ++i) acc[i].v = zero_v8f();

#if HAVE_TDM
    // Invariant parts of the Tensor DMA descriptor (D#).
    const unsigned lds_off = (unsigned)(unsigned long long)(uintptr_t)&ldsA[0];
    i32x8 g1;
    // dw0: data_size=1(2B)@16 | pad_enable@20 | pad_interval=3(16 DW)@22 | pad_amount=1(2 DW)@25
    g1[0] = (int)((1u << 16) | (1u << 20) | (3u << 22) | (1u << 25));
    g1[1] = (int)((CH & 0xFFFFu) << 16);                 // tensor_dim0[15:0] @ bits 63:48
    g1[2] = (int)((CH >> 16) | ((CH & 0xFFFFu) << 16));  // tensor_dim0 hi | tensor_dim1 lo
    g1[3] = (int)((CH >> 16) | (32u << 16));             // tensor_dim1 hi | tile_dim0=32 (K)
    g1[4] = (int)128;                                    // tile_dim1=128 rows, tile_dim2=0
    g1[5] = (int)CH;                                     // tensor_dim0_stride lo32
    g1[6] = 0;                                           // stride hi16 | dim1_stride lo16
    g1[7] = 0;
    i32x4 gz; gz[0] = 0; gz[1] = 0; gz[2] = 0; gz[3] = 0;
#endif

    for (int kc = 0; kc < CH; kc += 32) {
        __syncthreads();
#if HAVE_TDM
        if (wave == 0) {
            unsigned long long ga = (unsigned long long)(uintptr_t)(Wt + kc);
            u32x4 g0;
            g0[0] = 1u;             // count=1 valid descriptor
            g0[1] = lds_off;        // lds_addr (bytes)
            g0[2] = (unsigned)ga;   // global_addr[31:0]
            g0[3] = (unsigned)((ga >> 32) & 0x1FFFFFFu) | (2u << 30);  // addr[56:32] | type=2
#if __clang_major__ >= 23
            i32x8 gz8;
#pragma unroll
            for (int i = 0; i < 8; ++i) gz8[i] = 0;
            __builtin_amdgcn_tensor_load_to_lds(g0, g1, gz, gz, gz8, 0);
#else
            __builtin_amdgcn_tensor_load_to_lds(g0, g1, gz, gz, 0);
#endif
        }
#else
        // manual A staging: [128][36] padded rows, u32 pairs
#pragma unroll
        for (int i = 0; i < 8; ++i) {
            int e = t + i * 256;
            int m = e >> 4;
            int p = e & 15;
            *(unsigned int*)(ldsA + m * 36 + 2 * p) =
                *(const unsigned int*)(Wt + (size_t)m * CH + kc + 2 * p);
        }
#endif
        // stage B tile transposed: global [K][N] (u32 = 2 cols) -> LDS [N][K]
#pragma unroll
        for (int i = 0; i < 8; ++i) {
            int e  = t + i * 256;        // 2048 u32 elements
            int k  = e >> 6;             // 32 K rows
            int np = e & 63;             // column pair
            unsigned int xv =
                *(const unsigned int*)(Xb + (size_t)(kc + k) * SP + nblk * 128 + 2 * np);
            ldsB[(2 * np) * 34 + k]     = (unsigned short)xv;
            ldsB[(2 * np + 1) * 34 + k] = (unsigned short)(xv >> 16);
        }
        if (kc + 32 < CH) {
            __builtin_prefetch(Xb + (size_t)(kc + 32 + (t >> 7)) * SP + nblk * 128 + (t & 127), 0, 1);
        }
#if HAVE_TDM
        if (wave == 0) wait_tensorcnt0();
#endif
        __syncthreads();
        v16bf a = load_frag_a(ldsA, 36, wave * 16);
#pragma unroll
        for (int nst = 0; nst < 8; ++nst) {
            v16bf bb = load_frag_b(ldsB, 34, nst * 16);
            acc[nst].v = __builtin_amdgcn_wmma_f32_16x16x32_bf16(
                false, a, false, bb, (short)0, acc[nst].v, false, false);
        }
    }

    const int half8 = (lane < 16) ? 0 : 8;
    const int col0  = nblk * 128 + (lane & 15);
#pragma unroll
    for (int nst = 0; nst < 8; ++nst) {
        int ncol = col0 + nst * 16;
#pragma unroll
        for (int r = 0; r < 8; ++r) {
            int o       = mblk * 128 + wave * 16 + r + half8;
            float val   = acc[nst].f[r] + bias[o];
            size_t oidx = ((size_t)b * CH + o) * SP + ncol;
            if (MODE == 1) {
                float e = __expf(val);
                outb[oidx] = f2bf(e);
                float s = e;
                s += __shfl_xor(s, 1, 16);
                s += __shfl_xor(s, 2, 16);
                s += __shfl_xor(s, 4, 16);
                s += __shfl_xor(s, 8, 16);
                if ((lane & 15) == 0) atomicAdd(&ksum[b * CH + o], s);
            } else if (MODE == 3) {
                outf[oidx] = val;
            } else {
                outb[oidx] = f2bf(val);
            }
        }
    }
}

// ---------------- softmax over head_dim (d=32) per pixel, * d^-0.5 ----------------
__global__ void qsoftmax_kernel(unsigned short* __restrict__ qp) {
    size_t idx = (size_t)blockIdx.x * blockDim.x + threadIdx.x;  // B*NHD*SP threads
    int s  = (int)(idx % SP);
    int bn = (int)(idx / SP);
    int b  = bn >> 3;
    int n  = bn & 7;
    size_t base = ((size_t)b * CH + n * HD) * SP + s;
    float v[HD];
    float mx = -1e30f;
#pragma unroll
    for (int i = 0; i < HD; ++i) {
        v[i] = bf2f(qp[base + (size_t)i * SP]);
        mx   = fmaxf(mx, v[i]);
    }
    float sum = 0.0f;
#pragma unroll
    for (int i = 0; i < HD; ++i) {
        v[i] = __expf(v[i] - mx);
        sum += v[i];
    }
    float sc = 0.17677669529663687f / sum;  // 32^-0.5 / sum
#pragma unroll
    for (int i = 0; i < HD; ++i) qp[base + (size_t)i * SP] = f2bf(v[i] * sc);
}

// ---------------- ctx[i,j] += sum_s ekp[i,s]*vp[j,s], per (b,head), K split over waves ----------------
// Staging is a straight copy (both operands are [row][s] row-major), so use CDNA5
// async global->LDS loads (ASYNCcnt) when available: no VGPR round-trip, no barriers.
__global__ void ctx_kernel(const unsigned short* __restrict__ ekp,
                           const unsigned short* __restrict__ vp,
                           float* __restrict__ ctx) {
    __shared__ __align__(16) unsigned short lds[8][2 * 32 * 34];
    const int t = threadIdx.x, lane = t & 31, wave = t >> 5;
    const int bn = blockIdx.x;
    const size_t cbase = (size_t)bn * HD * SP;  // == ((b*CH)+n*HD)*SP
    unsigned short* lA = &lds[wave][0];
    unsigned short* lB = &lds[wave][32 * 34];

    AccF acc[2][2];
#pragma unroll
    for (int i = 0; i < 2; ++i)
#pragma unroll
        for (int j = 0; j < 2; ++j) acc[i][j].v = zero_v8f();

    const int kspan  = SP / 8;
    const int kstart = wave * kspan;
    for (int k0 = kstart; k0 < kstart + kspan; k0 += 32) {
#if HAVE_ASYNC
        // per-wave async staging: 512 b32 copies per 32x32 tile (16 per lane each)
#pragma unroll
        for (int i = 0; i < 16; ++i) {
            int e   = lane + i * 32;
            int row = e >> 4;
            int p   = e & 15;
            __builtin_amdgcn_global_load_async_to_lds_b32(
                TO_AS1(ekp + cbase + (size_t)row * SP + k0 + 2 * p),
                TO_AS3(lA + row * 34 + 2 * p), 0, 0);
            __builtin_amdgcn_global_load_async_to_lds_b32(
                TO_AS1(vp + cbase + (size_t)row * SP + k0 + 2 * p),
                TO_AS3(lB + row * 34 + 2 * p), 0, 0);
        }
        wait_asynccnt0();
#else
#pragma unroll
        for (int i = 0; i < 16; ++i) {
            int e   = lane + i * 32;
            int row = e >> 4;
            int p   = e & 15;
            *(unsigned int*)(lA + row * 34 + 2 * p) =
                *(const unsigned int*)(ekp + cbase + (size_t)row * SP + k0 + 2 * p);
            *(unsigned int*)(lB + row * 34 + 2 * p) =
                *(const unsigned int*)(vp + cbase + (size_t)row * SP + k0 + 2 * p);
        }
        __syncthreads();  // uniform trip count across all 8 waves
#endif
        v16bf a0 = load_frag_a(lA, 34, 0);
        v16bf a1 = load_frag_a(lA, 34, 16);
        v16bf b0 = load_frag_b(lB, 34, 0);
        v16bf b1 = load_frag_b(lB, 34, 16);
        acc[0][0].v = __builtin_amdgcn_wmma_f32_16x16x32_bf16(false, a0, false, b0, (short)0, acc[0][0].v, false, false);
        acc[0][1].v = __builtin_amdgcn_wmma_f32_16x16x32_bf16(false, a0, false, b1, (short)0, acc[0][1].v, false, false);
        acc[1][0].v = __builtin_amdgcn_wmma_f32_16x16x32_bf16(false, a1, false, b0, (short)0, acc[1][0].v, false, false);
        acc[1][1].v = __builtin_amdgcn_wmma_f32_16x16x32_bf16(false, a1, false, b1, (short)0, acc[1][1].v, false, false);
#if !HAVE_ASYNC
        __syncthreads();
#endif
    }
    const int half8 = (lane < 16) ? 0 : 8;
    const int col   = lane & 15;
#pragma unroll
    for (int mi = 0; mi < 2; ++mi)
#pragma unroll
        for (int nj = 0; nj < 2; ++nj)
#pragma unroll
            for (int r = 0; r < 8; ++r) {
                int i = mi * 16 + r + half8;
                int j = nj * 16 + col;
                atomicAdd(&ctx[((size_t)bn * HD + i) * HD + j], acc[mi][nj].f[r]);
            }
}

// ---------------- out[j,s] = sum_i (ctx[i,j]/ksum[i]) * qs[i,s], per (b,head,sblock) ----------------
__global__ void attnout_kernel(const float* __restrict__ ctx,
                               const float* __restrict__ ksum,
                               const unsigned short* __restrict__ qs,
                               unsigned short* __restrict__ attn) {
    __shared__ __align__(16) unsigned short lA[32 * 34];  // A' = (ctx/ksum)^T as [j][i]
    const int t = threadIdx.x, lane = t & 31, wave = t >> 5;
    const int bn   = blockIdx.y;
    const int b    = bn >> 3;
    const int n    = bn & 7;
    const int sblk = blockIdx.x;  // SP/256 blocks
    for (int e = t; e < 1024; e += 256) {
        int j = e >> 5, i = e & 31;
        float val = ctx[((size_t)bn * HD + i) * HD + j] / ksum[b * CH + n * HD + i];
        lA[j * 34 + i] = f2bf(val);
    }
    __syncthreads();
    const size_t cbase = ((size_t)b * CH + n * HD) * SP;
    const int kbase = (lane < 16) ? 0 : 16;
    const int half8 = (lane < 16) ? 0 : 8;
#pragma unroll
    for (int q = 0; q < 4; ++q) {
        int st  = wave + q * 8;       // 32 subtiles: 2 (M) x 16 (N)
        int mst = st & 1;
        int nst = st >> 1;
        int col = sblk * 256 + nst * 16 + (lane & 15);
        FragB bf;
#pragma unroll
        for (int v = 0; v < 8; ++v) {
            int k = kbase + 2 * v;
            unsigned int h0 = qs[cbase + (size_t)k * SP + col];
            unsigned int h1 = qs[cbase + (size_t)(k + 1) * SP + col];
            bf.u[v] = h0 | (h1 << 16);
        }
        v16bf a = load_frag_a(lA, 34, mst * 16);
        AccF acc;
        acc.v = zero_v8f();
        acc.v = __builtin_amdgcn_wmma_f32_16x16x32_bf16(false, a, false, bf.v, (short)0, acc.v, false, false);
#pragma unroll
        for (int r = 0; r < 8; ++r) {
            int j = mst * 16 + r + half8;
            attn[cbase + (size_t)j * SP + col] = f2bf(acc.f[r]);
        }
    }
}

extern "C" void kernel_launch(void* const* d_in, const int* in_sizes, int n_in,
                              void* d_out, int out_size, void* d_ws, size_t ws_size,
                              hipStream_t stream) {
    (void)in_sizes; (void)n_in; (void)out_size; (void)ws_size;
    const float* q  = (const float*)d_in[0];
    const float* wq = (const float*)d_in[1];
    const float* bq = (const float*)d_in[2];
    const float* wk = (const float*)d_in[3];
    const float* bk = (const float*)d_in[4];
    const float* wv = (const float*)d_in[5];
    const float* bv = (const float*)d_in[6];
    const float* wo = (const float*)d_in[7];
    const float* bo = (const float*)d_in[8];
    float* out = (float*)d_out;

    char* ws = (char*)d_ws;
    size_t off = 0;
    auto carve = [&](size_t bytes) -> void* {
        void* p = ws + off;
        off += (bytes + 255) & ~(size_t)255;
        return p;
    };
    const size_t NX = (size_t)BATCH * CH * SP;
    unsigned short* xb   = (unsigned short*)carve(NX * 2);
    unsigned short* qp   = (unsigned short*)carve(NX * 2);
    unsigned short* ekp  = (unsigned short*)carve(NX * 2);
    unsigned short* vpb  = (unsigned short*)carve(NX * 2);
    unsigned short* attn = (unsigned short*)carve(NX * 2);
    unsigned short* wqb  = (unsigned short*)carve((size_t)CH * CH * 2);
    unsigned short* wkb  = (unsigned short*)carve((size_t)CH * CH * 2);
    unsigned short* wvb  = (unsigned short*)carve((size_t)CH * CH * 2);
    unsigned short* wob  = (unsigned short*)carve((size_t)CH * CH * 2);
    float* ksum = (float*)carve((size_t)BATCH * CH * 4);
    float* ctx  = (float*)carve((size_t)BATCH * NHD * HD * HD * 4);

    dim3 blk(256);
    const size_t NW = (size_t)CH * CH;

    f32_to_bf16_kernel<<<dim3((unsigned)((NX / 4 + 255) / 256)), blk, 0, stream>>>(q, xb, NX);
    f32_to_bf16_kernel<<<dim3((unsigned)((NW / 4 + 255) / 256)), blk, 0, stream>>>(wq, wqb, NW);
    f32_to_bf16_kernel<<<dim3((unsigned)((NW / 4 + 255) / 256)), blk, 0, stream>>>(wk, wkb, NW);
    f32_to_bf16_kernel<<<dim3((unsigned)((NW / 4 + 255) / 256)), blk, 0, stream>>>(wv, wvb, NW);
    f32_to_bf16_kernel<<<dim3((unsigned)((NW / 4 + 255) / 256)), blk, 0, stream>>>(wo, wob, NW);
    zero_f32_kernel<<<dim3((BATCH * CH + 255) / 256), blk, 0, stream>>>(ksum, BATCH * CH);
    zero_f32_kernel<<<dim3((BATCH * NHD * HD * HD + 255) / 256), blk, 0, stream>>>(
        ctx, (size_t)BATCH * NHD * HD * HD);

    dim3 ggrid(SP / 128, CH / 128, BATCH);
    proj_gemm_kernel<0><<<ggrid, blk, 0, stream>>>(xb, wqb, bq, qp, nullptr, nullptr);
    proj_gemm_kernel<1><<<ggrid, blk, 0, stream>>>(xb, wkb, bk, ekp, nullptr, ksum);
    proj_gemm_kernel<2><<<ggrid, blk, 0, stream>>>(xb, wvb, bv, vpb, nullptr, nullptr);

    qsoftmax_kernel<<<dim3((BATCH * NHD * SP) / 256), blk, 0, stream>>>(qp);
    ctx_kernel<<<dim3(BATCH * NHD), blk, 0, stream>>>(ekp, vpb, ctx);
    attnout_kernel<<<dim3(SP / 256, BATCH * NHD), blk, 0, stream>>>(ctx, ksum, qp, attn);

    proj_gemm_kernel<3><<<ggrid, blk, 0, stream>>>(attn, wob, bo, nullptr, out, nullptr);
}
